// PNAS_46574625358331
// MI455X (gfx1250) — matrix operand order repaired and checked by reading, not compiled, not run on hardware
//
#include <hip/hip_runtime.h>
#include <math.h>

// Problem constants (from reference)
#define NN 10000L
#define EE 160000L
#define FF 64
#define DD 32
#define HH 128
// AVG_DEG_LOG = log(17)
#define INV_AVG_DEG_LOG (1.0f / 2.8332133440562162f)

typedef __attribute__((ext_vector_type(16))) _Float16 v16h;
typedef __attribute__((ext_vector_type(8)))  float    v8f;
typedef __attribute__((ext_vector_type(4)))  unsigned int u32x4;

union FragU { v16h h; u32x4 u[2]; };

// ---- WMMA fragment loaders -------------------------------------------------
// A (16x32 f16): lane half=l>>4, m=l&15. halfs 0..7 <-> K = k0+8*half+0..7,
// halfs 8..15 <-> K = k0+16+8*half+0..7  (per CDNA5 ISA 7.12.2 table).
static __device__ inline v16h load_afrag(const _Float16* __restrict__ row, int k0, int half) {
  FragU f;
  const _Float16* p = row + k0 + 8 * half;
  f.u[0] = *(const u32x4*)(p);
  f.u[1] = *(const u32x4*)(p + 16);
  return f.h;
}
// B fragments are pre-swizzled in global memory to be lane-contiguous:
// layout [chunk][ntile][lane][16 halfs]
static __device__ inline v16h load_bfrag(const _Float16* __restrict__ wf, long chunk, int ntile, int lane) {
  FragU f;
  const _Float16* p = wf + (((chunk * 8 + ntile) * 32 + lane) << 4);
  f.u[0] = *(const u32x4*)(p);
  f.u[1] = *(const u32x4*)(p + 8);
  return f.h;
}

// ---- atomic float min/max via int punning ---------------------------------
static __device__ inline void atomicMaxF(float* a, float v) {
  if (v >= 0.0f) atomicMax((int*)a, __float_as_int(v));
  else           atomicMin((unsigned int*)a, __float_as_uint(v));
}
static __device__ inline void atomicMinF(float* a, float v) {
  if (v >= 0.0f) atomicMin((int*)a, __float_as_int(v));
  else           atomicMax((unsigned int*)a, __float_as_uint(v));
}

// ============================================================================
// Generic row-major WMMA GEMM: out[M x 128] = A16[M x K] @ Wfrag + bias
// Epilogue: optional relu; optional residual (val = resid + 0.5*val);
// writes f32 and/or f16 outputs (width fixed to 128).
// ============================================================================
__global__ void pna_wmma_gemm_kernel(const _Float16* __restrict__ A, long M, int K,
                                     const _Float16* __restrict__ Wf,
                                     const float* __restrict__ bias,
                                     float* __restrict__ outF, _Float16* __restrict__ outH,
                                     const float* __restrict__ residF, int relu)
{
  int wave = threadIdx.x >> 5;
  int lane = threadIdx.x & 31;
  long tile = (long)blockIdx.x * (blockDim.x >> 5) + wave;
  if (tile >= (M >> 4)) return;
  int m = lane & 15, half = lane >> 4;
  const _Float16* row = A + (tile * 16 + m) * (long)K;
  v8f zero = {0.f, 0.f, 0.f, 0.f, 0.f, 0.f, 0.f, 0.f};
  v8f acc[8];
#pragma unroll
  for (int t = 0; t < 8; ++t) acc[t] = zero;
  int nc = K >> 5;
  for (int c = 0; c < nc; ++c) {
    v16h a = load_afrag(row, c << 5, half);
#pragma unroll
    for (int t = 0; t < 8; ++t) {
      v16h b = load_bfrag(Wf, c, t, lane);
      acc[t] = __builtin_amdgcn_wmma_f32_16x16x32_f16(false, a, false, b, (short)0, acc[t], false, false);
    }
  }
  long r0 = tile * 16;
#pragma unroll
  for (int t = 0; t < 8; ++t) {
    int col = t * 16 + m;
    float bv = bias ? bias[col] : 0.f;
#pragma unroll
    for (int v = 0; v < 8; ++v) {
      long idx = (r0 + v + 8 * half) * 128 + col;
      float val = acc[t][v] + bv;
      if (relu) val = fmaxf(val, 0.f);
      if (residF) val = residF[idx] + 0.5f * val;
      if (outF) outF[idx] = val;
      if (outH) outH[idx] = (_Float16)val;
    }
  }
}

// ============================================================================
// Gather WMMA GEMM over edges: out[E x 128] =
//   [ x16[idx0[j]] | x16[idx1[j]] | e16[j] ] (E x 384) @ Wfrag + bias
// ============================================================================
__global__ void pna_wmma_gather_kernel(const _Float16* __restrict__ x16,
                                       const _Float16* __restrict__ e16,
                                       const int* __restrict__ idx0,
                                       const int* __restrict__ idx1,
                                       long M,
                                       const _Float16* __restrict__ Wf,
                                       const float* __restrict__ bias,
                                       float* __restrict__ outF, _Float16* __restrict__ outH,
                                       int relu)
{
  int wave = threadIdx.x >> 5;
  int lane = threadIdx.x & 31;
  long tile = (long)blockIdx.x * (blockDim.x >> 5) + wave;
  if (tile >= (M >> 4)) return;
  int m = lane & 15, half = lane >> 4;
  long j = tile * 16 + m;
  long i0 = idx0[j], i1 = idx1[j];
  const _Float16* r0p = x16 + i0 * 128;
  const _Float16* r1p = x16 + i1 * 128;
  const _Float16* r2p = e16 + j * 128;
  v8f zero = {0.f, 0.f, 0.f, 0.f, 0.f, 0.f, 0.f, 0.f};
  v8f acc[8];
#pragma unroll
  for (int t = 0; t < 8; ++t) acc[t] = zero;
  for (int c = 0; c < 12; ++c) {             // K = 384 = 12 chunks of 32
    int k0 = c << 5;
    const _Float16* row;
    int kk;
    if (k0 < 128)      { row = r0p; kk = k0;       }
    else if (k0 < 256) { row = r1p; kk = k0 - 128; }
    else               { row = r2p; kk = k0 - 256; }
    v16h a = load_afrag(row, kk, half);
#pragma unroll
    for (int t = 0; t < 8; ++t) {
      v16h b = load_bfrag(Wf, c, t, lane);
      acc[t] = __builtin_amdgcn_wmma_f32_16x16x32_f16(false, a, false, b, (short)0, acc[t], false, false);
    }
  }
  long rb = tile * 16;
#pragma unroll
  for (int t = 0; t < 8; ++t) {
    int col = t * 16 + m;
    float bv = bias ? bias[col] : 0.f;
#pragma unroll
    for (int v = 0; v < 8; ++v) {
      long idx = (rb + v + 8 * half) * 128 + col;
      float val = acc[t][v] + bv;
      if (relu) val = fmaxf(val, 0.f);
      if (outF) outF[idx] = val;
      if (outH) outH[idx] = (_Float16)val;
    }
  }
}

// ============================================================================
// Setup / pointwise kernels
// ============================================================================
__global__ void pna_cvt_f16_kernel(const float* __restrict__ in, _Float16* __restrict__ out, long n) {
  long i = (long)blockIdx.x * blockDim.x + threadIdx.x;
  if (i < n) out[i] = (_Float16)in[i];
}

// Swizzle f32 KxH weight into fragment-ordered f16: [chunk][ntile][lane][16]
__global__ void pna_swizzle_w_kernel(const float* __restrict__ W, int K, _Float16* __restrict__ out) {
  long gid = (long)blockIdx.x * blockDim.x + threadIdx.x;
  if (gid >= (long)K * 128) return;
  int k = (int)(gid >> 7), n = (int)(gid & 127);
  int chunk = k >> 5, kk = k & 31;
  int lane = (n & 15) | (((kk >> 3) & 1) << 4);
  int ntile = n >> 4, i = (kk & 7) | ((kk >> 4) << 3);
  out[(((long)(chunk * 8 + ntile)) * 32 + lane) * 16 + i] = (_Float16)W[gid];
}

// Wmsg = [Wpre[0:2H] ; Wee @ Wpre[2H:3H]]   (384 x 128, f32)
__global__ void pna_build_wmsg_kernel(const float* __restrict__ Wpre, const float* __restrict__ Wee,
                                      float* __restrict__ out) {
  long gid = (long)blockIdx.x * blockDim.x + threadIdx.x;
  if (gid >= 384L * 128) return;
  int r = (int)(gid >> 7), c = (int)(gid & 127);
  if (r < 256) { out[gid] = Wpre[gid]; return; }
  float s = 0.f;
  const float* wr = Wee + (long)(r - 256) * 128;
  for (int k = 0; k < 128; ++k) s += wr[k] * Wpre[(long)(256 + k) * 128 + c];
  out[gid] = s;
}
__global__ void pna_build_bmsg_kernel(const float* __restrict__ bpre, const float* __restrict__ bee,
                                      const float* __restrict__ Wpre, float* __restrict__ out) {
  int c = threadIdx.x;
  float s = bpre[c];
  for (int k = 0; k < 128; ++k) s += bee[k] * Wpre[(long)(256 + k) * 128 + c];
  out[c] = s;
}
// Wpostc = W_post @ W_lin  (1664 x 128, f32)
__global__ void pna_build_wpost_kernel(const float* __restrict__ Wp, const float* __restrict__ Wl,
                                       float* __restrict__ out) {
  long gid = (long)blockIdx.x * blockDim.x + threadIdx.x;
  if (gid >= 1664L * 128) return;
  int r = (int)(gid >> 7), c = (int)(gid & 127);
  float s = 0.f;
  const float* wr = Wp + (long)r * 128;
  for (int k = 0; k < 128; ++k) s += wr[k] * Wl[(long)k * 128 + c];
  out[gid] = s;
}
__global__ void pna_build_bpost_kernel(const float* __restrict__ bpost, const float* __restrict__ blin,
                                       const float* __restrict__ Wl, float* __restrict__ out) {
  int c = threadIdx.x;
  float s = blin[c];
  for (int k = 0; k < 128; ++k) s += bpost[k] * Wl[(long)k * 128 + c];
  out[c] = s;
}

// ============================================================================
// Aggregation / BN kernels
// ============================================================================
__global__ void pna_agg_init_kernel(float* sum, float* sq, float* mn, float* mx, float* cnt, long n) {
  long i = (long)blockIdx.x * blockDim.x + threadIdx.x;
  if (i < n) { sum[i] = 0.f; sq[i] = 0.f; mn[i] = __builtin_inff(); mx[i] = -__builtin_inff(); }
  if (i < NN) cnt[i] = 0.f;
}
__global__ void pna_agg_scatter_kernel(const float* __restrict__ h, const int* __restrict__ dst,
                                       float* sum, float* sq, float* mn, float* mx, float* cnt) {
  long gid = (long)blockIdx.x * blockDim.x + threadIdx.x;
  if (gid >= EE * 128) return;
  long j = gid >> 7; int c = (int)(gid & 127);
  float v = h[gid];
  int d = dst[j];
  long b = (long)d * 128 + c;
  atomicAdd(sum + b, v);
  atomicAdd(sq + b, v * v);
  atomicMaxF(mx + b, v);
  atomicMinF(mn + b, v);
  if (c == 0) atomicAdd(cnt + d, 1.0f);
}
// Build nodecat (N x 13H, f16): [x | mean,mn,mx,std | *f | /f]
__global__ void pna_nodecat_kernel(const float* __restrict__ xF,
                                   const float* __restrict__ sum, const float* __restrict__ sq,
                                   const float* __restrict__ mn, const float* __restrict__ mx,
                                   const float* __restrict__ cnt, _Float16* __restrict__ out) {
  long gid = (long)blockIdx.x * blockDim.x + threadIdx.x;
  if (gid >= NN * 128) return;
  long n = gid >> 7; int c = (int)(gid & 127);
  float cn = cnt[n];
  float degc = fmaxf(cn, 1.0f);
  float mean = sum[gid] / degc;
  float msq  = sq[gid] / degc;
  float stdv = sqrtf(fmaxf(msq - mean * mean, 0.f) + 1e-5f);
  float vmn = (cn > 0.f) ? mn[gid] : 0.f;
  float vmx = (cn > 0.f) ? mx[gid] : 0.f;
  float f = logf(degc + 1.f) * INV_AVG_DEG_LOG;
  float invf = 1.f / f;
  _Float16* row = out + n * 1664 + c;
  row[0]    = (_Float16)xF[gid];
  row[128]  = (_Float16)mean;        row[256]  = (_Float16)vmn;
  row[384]  = (_Float16)vmx;         row[512]  = (_Float16)stdv;
  row[640]  = (_Float16)(mean * f);  row[768]  = (_Float16)(vmn * f);
  row[896]  = (_Float16)(vmx * f);   row[1024] = (_Float16)(stdv * f);
  row[1152] = (_Float16)(mean*invf); row[1280] = (_Float16)(vmn * invf);
  row[1408] = (_Float16)(vmx*invf);  row[1536] = (_Float16)(stdv * invf);
}
__global__ void pna_zero_kernel(float* p, int n) {
  int i = blockIdx.x * blockDim.x + threadIdx.x;
  if (i < n) p[i] = 0.f;
}
// Column sums/sumsq for BatchNorm (blockDim.x == 128, one col per thread)
__global__ void pna_bn_stats_kernel(const float* __restrict__ o, float* __restrict__ bnstat) {
  int c = threadIdx.x;
  long r0 = (long)blockIdx.x * 64;
  float s = 0.f, q = 0.f;
  for (int r = 0; r < 64; ++r) {
    long row = r0 + r;
    if (row < NN) { float v = o[row * 128 + c]; s += v; q += v * v; }
  }
  atomicAdd(bnstat + c, s);
  atomicAdd(bnstat + 128 + c, q);
}
// bn -> relu -> x=(x+relu(bn))/2, refresh f16 copy
__global__ void pna_bn_apply_kernel(const float* __restrict__ o, const float* __restrict__ bnstat,
                                    const float* __restrict__ gamma, const float* __restrict__ beta,
                                    float* __restrict__ xF, _Float16* __restrict__ xh16) {
  long gid = (long)blockIdx.x * blockDim.x + threadIdx.x;
  if (gid >= NN * 128) return;
  int c = (int)(gid & 127);
  float mu  = bnstat[c] * (1.0f / (float)NN);
  float var = bnstat[128 + c] * (1.0f / (float)NN) - mu * mu;
  float bn = gamma[c] * (o[gid] - mu) * rsqrtf(var + 1e-5f) + beta[c];
  float xn = (xF[gid] + fmaxf(bn, 0.f)) * 0.5f;
  xF[gid] = xn;
  xh16[gid] = (_Float16)xn;
}

// ============================================================================
static inline int cdiv(long a, long b) { return (int)((a + b - 1) / b); }

extern "C" void kernel_launch(void* const* d_in, const int* in_sizes, int n_in,
                              void* d_out, int out_size, void* d_ws, size_t ws_size,
                              hipStream_t stream) {
  (void)in_sizes; (void)n_in; (void)out_size; (void)ws_size;
  const float* x      = (const float*)d_in[0];
  const int*   eidx   = (const int*)d_in[1];
  const float* eattr  = (const float*)d_in[2];
  const float* W_node = (const float*)d_in[3];  const float* b_node = (const float*)d_in[4];
  const float* W_edge = (const float*)d_in[5];  const float* b_edge = (const float*)d_in[6];
  const float* W_ee   = (const float*)d_in[7];  const float* b_ee   = (const float*)d_in[8];
  const float* W_pre  = (const float*)d_in[9];  const float* b_pre  = (const float*)d_in[10];
  const float* W_post = (const float*)d_in[11]; const float* b_post = (const float*)d_in[12];
  const float* W_lin  = (const float*)d_in[13]; const float* b_lin  = (const float*)d_in[14];
  const float* gamma  = (const float*)d_in[15]; const float* beta   = (const float*)d_in[16];
  const float* We1    = (const float*)d_in[17]; const float* be1    = (const float*)d_in[18];
  const float* We2    = (const float*)d_in[19]; const float* be2    = (const float*)d_in[20];
  const int* src = eidx;          // edge_index[0]
  const int* dst = eidx + EE;     // edge_index[1]

  float* xF = (float*)d_out;          // live node state (N x 128)
  float* eF = xF + NN * 128;          // live edge state (E x 128)

  // ---- workspace carve-out ----
  char* p = (char*)d_ws;
  auto alloc = [&](size_t bytes) -> char* {
    char* r = p; p += (bytes + 255) & ~(size_t)255; return r;
  };
  _Float16* x16   = (_Float16*)alloc(NN * FF * 2);
  _Float16* ea16  = (_Float16*)alloc(EE * DD * 2);
  _Float16* xh16  = (_Float16*)alloc(NN * 128 * 2);
  _Float16* e16   = (_Float16*)alloc(EE * 128 * 2);
  float*    h     = (float*)   alloc(EE * 128 * 4);      // 81.92 MB, aliased below
  _Float16* nodecat16 = (_Float16*)h;                    // N*1664*2 = 33.28 MB
  _Float16* t16   = (_Float16*)((char*)h + NN * 1664 * 2); // E*128*2 = 40.96 MB (fits)
  float* aggsum = (float*)alloc(NN * 128 * 4);
  float* aggsq  = (float*)alloc(NN * 128 * 4);
  float* aggmn  = (float*)alloc(NN * 128 * 4);
  float* aggmx  = (float*)alloc(NN * 128 * 4);
  float* cnt    = (float*)alloc(NN * 4);
  float* onode  = (float*)alloc(NN * 128 * 4);
  float* bnstat = (float*)alloc(2 * 128 * 4);
  float* wtmp   = (float*)alloc(1664L * 128 * 4);
  _Float16* Wmsg_f  = (_Float16*)alloc(384L * 128 * 2);
  _Float16* Wpost_f = (_Float16*)alloc(1664L * 128 * 2);
  _Float16* We1_f   = (_Float16*)alloc(384L * 128 * 2);
  _Float16* We2_f   = (_Float16*)alloc(128L * 128 * 2);
  _Float16* Wnode_f = (_Float16*)alloc(64L * 128 * 2);
  _Float16* Wedge_f = (_Float16*)alloc(32L * 128 * 2);
  float* bmsg  = (float*)alloc(128 * 4);
  float* bpost = (float*)alloc(128 * 4);

  const int BT = 256;

  // ---- convert inputs + initial encoders ----
  pna_cvt_f16_kernel<<<cdiv(NN * FF, BT), BT, 0, stream>>>(x, x16, NN * FF);
  pna_cvt_f16_kernel<<<cdiv(EE * DD, BT), BT, 0, stream>>>(eattr, ea16, EE * DD);
  pna_swizzle_w_kernel<<<cdiv(64L * 128, BT), BT, 0, stream>>>(W_node, 64, Wnode_f);
  pna_swizzle_w_kernel<<<cdiv(32L * 128, BT), BT, 0, stream>>>(W_edge, 32, Wedge_f);
  // xh = x @ W_node + b ;  e = edge_attr @ W_edge + b   (both f32 + f16 shadow)
  pna_wmma_gemm_kernel<<<cdiv(NN / 16, 4), 128, 0, stream>>>(x16, NN, 64, Wnode_f, b_node, xF, xh16, nullptr, 0);
  pna_wmma_gemm_kernel<<<cdiv(EE / 16, 4), 128, 0, stream>>>(ea16, EE, 32, Wedge_f, b_edge, eF, e16, nullptr, 0);

  for (int l = 0; l < 2; ++l) {
    const float* Wpre_l  = W_pre  + (long)l * 384 * 128;
    const float* Wee_l   = W_ee   + (long)l * 128 * 128;
    const float* Wpost_l = W_post + (long)l * 1664 * 128;
    const float* Wlin_l  = W_lin  + (long)l * 128 * 128;
    const float* We1_l   = We1    + (long)l * 384 * 128;
    const float* We2_l   = We2    + (long)l * 128 * 128;

    // --- fold + swizzle weights for this layer ---
    pna_build_wmsg_kernel<<<cdiv(384L * 128, BT), BT, 0, stream>>>(Wpre_l, Wee_l, wtmp);
    pna_swizzle_w_kernel<<<cdiv(384L * 128, BT), BT, 0, stream>>>(wtmp, 384, Wmsg_f);
    pna_build_bmsg_kernel<<<1, 128, 0, stream>>>(b_pre + l * 128, b_ee + l * 128, Wpre_l, bmsg);
    pna_build_wpost_kernel<<<cdiv(1664L * 128, BT), BT, 0, stream>>>(Wpost_l, Wlin_l, wtmp);
    pna_swizzle_w_kernel<<<cdiv(1664L * 128, BT), BT, 0, stream>>>(wtmp, 1664, Wpost_f);
    pna_build_bpost_kernel<<<1, 128, 0, stream>>>(b_post + l * 128, b_lin + l * 128, Wlin_l, bpost);
    pna_swizzle_w_kernel<<<cdiv(384L * 128, BT), BT, 0, stream>>>(We1_l, 384, We1_f);
    pna_swizzle_w_kernel<<<cdiv(128L * 128, BT), BT, 0, stream>>>(We2_l, 128, We2_f);

    // --- messages: h = [x_dst | x_src | e] @ Wmsg + bmsg  (f32) ---
    pna_wmma_gather_kernel<<<cdiv(EE / 16, 4), 128, 0, stream>>>(xh16, e16, dst, src, EE, Wmsg_f, bmsg, h, nullptr, 0);

    // --- multi-aggregator scatter at dst ---
    pna_agg_init_kernel<<<cdiv(NN * 128, BT), BT, 0, stream>>>(aggsum, aggsq, aggmn, aggmx, cnt, NN * 128);
    pna_agg_scatter_kernel<<<cdiv(EE * 128, BT), BT, 0, stream>>>(h, dst, aggsum, aggsq, aggmn, aggmx, cnt);

    // --- nodecat (f16) -> fused post GEMM (W_post @ W_lin folded) ---
    pna_nodecat_kernel<<<cdiv(NN * 128, BT), BT, 0, stream>>>(xF, aggsum, aggsq, aggmn, aggmx, cnt, nodecat16);
    pna_wmma_gemm_kernel<<<cdiv(NN / 16, 4), 128, 0, stream>>>(nodecat16, NN, 1664, Wpost_f, bpost, onode, nullptr, nullptr, 0);

    // --- BatchNorm + residual node update ---
    pna_zero_kernel<<<1, 256, 0, stream>>>(bnstat, 256);
    pna_bn_stats_kernel<<<cdiv(NN, 64), 128, 0, stream>>>(onode, bnstat);
    pna_bn_apply_kernel<<<cdiv(NN * 128, BT), BT, 0, stream>>>(onode, bnstat, gamma + l * 128, beta + l * 128, xF, xh16);

    // --- edge update: t = relu([x_src | x_dst | e] @ We1 + be1); e += 0.5*(t @ We2 + be2) ---
    pna_wmma_gather_kernel<<<cdiv(EE / 16, 4), 128, 0, stream>>>(xh16, e16, src, dst, EE, We1_f, be1 + l * 128, nullptr, t16, 1);
    pna_wmma_gemm_kernel<<<cdiv(EE / 16, 4), 128, 0, stream>>>(t16, EE, 128, We2_f, be2 + l * 128, eF, e16, eF, 0);
  }
  // d_out already holds [xh (N x 128) | e (E x 128)] in fp32.
}
